// QuantumKernelMethod_65481071403914
// MI455X (gfx1250) — compile-verified, matrix operand order checked by reading, not run on hardware
//
#include <hip/hip_runtime.h>
#include <hip/hip_bf16.h>
#include <math.h>

// ---------------------------------------------------------------------------
// QuantumKernelMethod on MI455X (gfx1250).
//
// Closed-form reduction: RZ layers (shared params) contribute only a fixed
// per-basis phase, CNOT ladders only a fixed basis permutation; both cancel
// in |<psi_x|psi_y>|^2.  The problem is exactly:
//     out[x][y] = ( sum_b R_x[b] * R_y[b] )^2
//     R_x[b]    = prod_i q^x_i[bit_i(b)],  q = (cos(x_i/2), sin(x_i/2))
// i.e. a real 4096 x 4096 x 1024 GEMM of product-state feature vectors.
// We scale each factor by sqrt(2) (so sum_b R'^2 = 1024, f16-friendly) and
// divide the dot product by 1024 before squaring.
// ---------------------------------------------------------------------------

typedef __attribute__((ext_vector_type(16))) _Float16 v16h;
typedef __attribute__((ext_vector_type(8)))  _Float16 v8h;
typedef __attribute__((ext_vector_type(8)))  float    v8f;

#define NSAMP   4096
#define NWIRES  10
#define DIM     1024            // 2^NWIRES
#define SQRT2   1.41421356237309504880f

// ---------------------------------------------------------------------------
// Phase 1: feature vectors  R'_x[b] = prod_i sqrt(2)*q^x_i[bit_i(b)]  -> f16
// One block per sample (blocks [0,4096) -> X, [4096,8192) -> Y).
// ---------------------------------------------------------------------------
__global__ void __launch_bounds__(256)
features_kernel(const float* __restrict__ X, const float* __restrict__ Y,
                _Float16* __restrict__ FX, _Float16* __restrict__ FY)
{
    const int s = blockIdx.x;
    const float* src;
    _Float16*    dst;
    if (s < NSAMP) { src = X + (size_t)s * NWIRES;           dst = FX + (size_t)s * DIM; }
    else           { src = Y + (size_t)(s - NSAMP) * NWIRES; dst = FY + (size_t)(s - NSAMP) * DIM; }

    __shared__ float qc[NWIRES];   // sqrt(2)*cos(x_i/2)
    __shared__ float qs[NWIRES];   // sqrt(2)*sin(x_i/2)
    if (threadIdx.x < NWIRES) {
        float h = 0.5f * src[threadIdx.x];
        qc[threadIdx.x] = SQRT2 * __cosf(h);
        qs[threadIdx.x] = SQRT2 * __sinf(h);
    }
    __syncthreads();

#pragma unroll
    for (int t = 0; t < DIM / 256; ++t) {
        const int b = threadIdx.x + t * 256;
        float p = 1.0f;
#pragma unroll
        for (int i = 0; i < NWIRES; ++i) {
            const int bit = (b >> (NWIRES - 1 - i)) & 1;
            p *= bit ? qs[i] : qc[i];
        }
        dst[b] = (_Float16)p;
    }
}

// ---------------------------------------------------------------------------
// Fragment loader for v_wmma_f32_16x16x32_f16 operands.
// 16-bit A layout (CDNA5 ISA 7.12.2), row-major source with leading dim DIM:
//   lanes 0-15 : row = base+lane,    K = {k..k+7}   and {k+16..k+23}
//   lanes 16-31: row = base+lane-16, K = {k+8..k+15} and {k+24..k+31}
// B of C = FX*FY^T is FY^T, so B columns are FY rows: same loader for both.
// ---------------------------------------------------------------------------
__device__ __forceinline__ v16h load_frag(const _Float16* __restrict__ base,
                                          int row_base, int k_base, int lane)
{
    const int row = row_base + (lane & 15);
    const int k0  = k_base + ((lane & 16) ? 8 : 0);
    const _Float16* p = base + (size_t)row * DIM + k0;
    const v8h lo = *(const v8h*)(p);        // K = k0 .. k0+7      (b128 load)
    const v8h hi = *(const v8h*)(p + 16);   // K = k0+16 .. k0+23  (b128 load)
    v16h out;
#pragma unroll
    for (int j = 0; j < 8; ++j) { out[j] = lo[j]; out[j + 8] = hi[j]; }
    return out;
}

// ---------------------------------------------------------------------------
// Phase 2: out = ((FX * FY^T)/1024)^2 via WMMA f16 -> f32.
// 256 threads = 8 wave32 waves. Block tile 128x128; each wave owns a 64x32
// tile as a 4x2 grid of 16x16 accumulators (64 acc VGPRs/lane).
// K = 1024 -> 32 steps of v_wmma_f32_16x16x32_f16 (8 WMMAs per step).
// ---------------------------------------------------------------------------
__global__ void __launch_bounds__(256)
qk_gemm_wmma(const _Float16* __restrict__ FX, const _Float16* __restrict__ FY,
             float* __restrict__ out)
{
    const int lane    = threadIdx.x & 31;
    const int wave    = threadIdx.x >> 5;          // 0..7
    const int waveM   = (wave >> 2) * 64;          // 2 waves along M
    const int waveN   = (wave & 3) * 32;           // 4 waves along N
    const int blockM  = blockIdx.y * 128;
    const int blockN  = blockIdx.x * 128;

    const int mBase = blockM + waveM;              // this wave's A-row base
    const int nBase = blockN + waveN;              // this wave's B-row base (FY rows)

    v8f acc[4][2];
#pragma unroll
    for (int mi = 0; mi < 4; ++mi)
#pragma unroll
        for (int ni = 0; ni < 2; ++ni)
            acc[mi][ni] = (v8f){0.f, 0.f, 0.f, 0.f, 0.f, 0.f, 0.f, 0.f};

    for (int k = 0; k < DIM; k += 32) {
        v16h a[4], b[2];
#pragma unroll
        for (int mi = 0; mi < 4; ++mi) a[mi] = load_frag(FX, mBase + mi * 16, k, lane);
#pragma unroll
        for (int ni = 0; ni < 2; ++ni) b[ni] = load_frag(FY, nBase + ni * 16, k, lane);

#pragma unroll
        for (int mi = 0; mi < 4; ++mi)
#pragma unroll
            for (int ni = 0; ni < 2; ++ni)
                acc[mi][ni] = __builtin_amdgcn_wmma_f32_16x16x32_f16(
                    /*neg_a=*/false, a[mi], /*neg_b=*/false, b[ni],
                    /*c_mod=*/(short)0, acc[mi][ni],
                    /*reuse_a=*/false, /*reuse_b=*/false);
    }

    // Store: C/D layout -> VGPR r holds M = r (lanes 0-15) / r+8 (lanes 16-31),
    // N = lane & 15.  Apply (dot/1024)^2.
    const float inv = 1.0f / 1024.0f;
    const int rowOff = (lane & 16) ? 8 : 0;
    const int col0   = lane & 15;
#pragma unroll
    for (int mi = 0; mi < 4; ++mi) {
#pragma unroll
        for (int ni = 0; ni < 2; ++ni) {
#pragma unroll
            for (int r = 0; r < 8; ++r) {
                const int row = mBase + mi * 16 + rowOff + r;
                const int col = nBase + ni * 16 + col0;
                const float v = acc[mi][ni][r] * inv;
                out[(size_t)row * NSAMP + col] = v * v;
            }
        }
    }
}

// ---------------------------------------------------------------------------
// Harness entry point.  d_in[0]=X (4096x10 f32), d_in[1]=Y (4096x10 f32),
// d_in[2]=params (unused: provably cancels).  d_out = 4096x4096 f32.
// Workspace: FX (8 MB f16) + FY (8 MB f16).
// ---------------------------------------------------------------------------
extern "C" void kernel_launch(void* const* d_in, const int* in_sizes, int n_in,
                              void* d_out, int out_size, void* d_ws, size_t ws_size,
                              hipStream_t stream)
{
    const float* X = (const float*)d_in[0];
    const float* Y = (const float*)d_in[1];
    (void)in_sizes; (void)n_in; (void)out_size; (void)ws_size;

    _Float16* FX = (_Float16*)d_ws;
    _Float16* FY = FX + (size_t)NSAMP * DIM;
    float*    O  = (float*)d_out;

    features_kernel<<<2 * NSAMP, 256, 0, stream>>>(X, Y, FX, FY);

    dim3 grid(NSAMP / 128, NSAMP / 128);   // 32 x 32 blocks
    qk_gemm_wmma<<<grid, 256, 0, stream>>>(FX, FY, O);
}
